// TSAEncoder_31903017075245
// MI455X (gfx1250) — compile-verified, matrix operand order checked by reading, not run on hardware
//
#include <hip/hip_runtime.h>
#include <cmath>

typedef __bf16 bf16;
typedef __attribute__((ext_vector_type(16))) __bf16 v16bf;
typedef __attribute__((ext_vector_type(8)))  float  v8f;
typedef unsigned int u32x4 __attribute__((ext_vector_type(4)));
typedef int i32x4e __attribute__((ext_vector_type(4)));
typedef int i32x8e __attribute__((ext_vector_type(8)));

union AFrag { v16bf v; u32x4 q[2]; };

__device__ __forceinline__ float gelu_exact(float v) {
  return 0.5f * v * (1.0f + erff(v * 0.70710678118654752f));
}

// Low 32 bits of a generic pointer to __shared__ = LDS byte address (ISA 10.2).
__device__ __forceinline__ unsigned lds_lo32(const void* p) {
  return (unsigned)(unsigned long long)(uintptr_t)p;
}

// ---------------------------------------------------------------------------
// Tensor Data Mover: DMA a 2D bf16 tile [tileD1 rows x tileD0 elems] from a
// row-major tensor (row stride strideD0 elems) into LDS (contiguous rows).
// Rows/cols beyond tensorD0/tensorD1 are zero-filled by TDM OOB handling.
// D# bitfields per CDNA5 ISA (group0 128b, group1 256b).
// ---------------------------------------------------------------------------
__device__ __forceinline__ void tdm_load_tile_2d(
    unsigned lds_byte_addr, const bf16* gtile,
    unsigned tensorD0, unsigned tensorD1,
    unsigned tileD0, unsigned tileD1, unsigned long long strideD0) {
  unsigned long long ga = (unsigned long long)(uintptr_t)gtile;
  u32x4 g0;
  g0[0] = 1u;                                                   // count=1, user mode
  g0[1] = lds_byte_addr;                                        // lds_addr
  g0[2] = (unsigned)(ga & 0xFFFFFFFFull);                       // global_addr[31:0]
  g0[3] = (unsigned)((ga >> 32) & 0x01FFFFFFull) | (2u << 30);  // [56:32] + type=2
  i32x8e g1;
  g1[0] = (int)(1u << 16);                                      // data_size=1 (2B)
  g1[1] = (int)((tensorD0 & 0xFFFFu) << 16);                    // tensor_dim0 lo16
  g1[2] = (int)(((tensorD0 >> 16) & 0xFFFFu) | ((tensorD1 & 0xFFFFu) << 16));
  g1[3] = (int)(((tensorD1 >> 16) & 0xFFFFu) | ((tileD0 & 0xFFFFu) << 16));
  g1[4] = (int)(tileD1 & 0xFFFFu);                              // tile_dim1, tile_dim2=0
  g1[5] = (int)(unsigned)(strideD0 & 0xFFFFFFFFull);            // dim0_stride lo32
  g1[6] = (int)(unsigned)((strideD0 >> 32) & 0xFFFFull);        // dim0_stride hi16
  g1[7] = 0;
  i32x4e z4 = {0, 0, 0, 0};
#if defined(__clang_major__) && (__clang_major__ >= 23)
  i32x8e z8 = {0, 0, 0, 0, 0, 0, 0, 0};
  __builtin_amdgcn_tensor_load_to_lds(g0, g1, z4, z4, z8, 0);
#else
  __builtin_amdgcn_tensor_load_to_lds(g0, g1, z4, z4, 0);
#endif
}

// ---------------------------------------------------------------------------
// Weight convert + transpose: W[K][N] f32 -> Wt[N][K] bf16
// ---------------------------------------------------------------------------
__global__ void wcvt_k(const float* __restrict__ W, bf16* __restrict__ Wt, int K, int N) {
  int i = blockIdx.x * 256 + threadIdx.x;
  if (i < K * N) {
    int k = i / N, n = i % N;
    Wt[(size_t)n * K + k] = (bf16)W[i];
  }
}

// router [32,10,512] f32 -> padded [32*16, 512] bf16 (rows f>=10 zeroed)
__global__ void router_pad_k(const float* __restrict__ R, bf16* __restrict__ out) {
  int i = blockIdx.x * 256 + threadIdx.x;       // 32*16*512 = 262144
  int d = i & 511, f = (i >> 9) & 15, s = i >> 13;
  out[i] = (f < 10) ? (bf16)R[(size_t)(s * 10 + f) * 512 + d] : (bf16)0.0f;
}

// ---------------------------------------------------------------------------
// SegMerging gather + LayerNorm over WIN*D=1024, emit bf16 GEMM operand
// ---------------------------------------------------------------------------
__global__ __launch_bounds__(256) void merge_ln_k(
    const float* __restrict__ x, const float* __restrict__ g,
    const float* __restrict__ be, bf16* __restrict__ out) {
  const int row = blockIdx.x;                   // bc*32 + s
  const int bc = row >> 5, sIdx = row & 31;
  const int tid = threadIdx.x;
  float v[4]; float s1 = 0.f, s2 = 0.f;
#pragma unroll
  for (int t = 0; t < 4; ++t) {
    int j = tid + t * 256;
    int i = j >> 9, d = j & 511;                // xm[.., i*D+d] = x[bc, s*WIN+i, d]
    float val = x[(size_t)(bc * 64 + sIdx * 2 + i) * 512 + d];
    v[t] = val; s1 += val; s2 += val * val;
  }
  __shared__ float r1[256], r2[256];
  r1[tid] = s1; r2[tid] = s2;
  __syncthreads();
  for (int st = 128; st > 0; st >>= 1) {
    if (tid < st) { r1[tid] += r1[tid + st]; r2[tid] += r2[tid + st]; }
    __syncthreads();
  }
  float mean = r1[0] * (1.f / 1024.f);
  float var  = r2[0] * (1.f / 1024.f) - mean * mean;
  float rinv = rsqrtf(var + 1e-5f);
#pragma unroll
  for (int t = 0; t < 4; ++t) {
    int j = tid + t * 256;
    out[(size_t)row * 1024 + j] = (bf16)((v[t] - mean) * rinv * g[j] + be[j]);
  }
}

// ---------------------------------------------------------------------------
// LayerNorm over D=512; optional permuted output row (swap two 5-bit fields:
// (b*32+i)*32+j -> (b*32+j)*32+i) used for t->ds and ds->out reshuffles.
// ---------------------------------------------------------------------------
__global__ __launch_bounds__(256) void ln512_k(
    const float* __restrict__ in, const float* __restrict__ g,
    const float* __restrict__ be, float* __restrict__ outF,
    bf16* __restrict__ outB, int permute) {
  const int r = blockIdx.x, tid = threadIdx.x;
  const size_t base = (size_t)r * 512;
  float v0 = in[base + tid * 2];
  float v1 = in[base + tid * 2 + 1];
  __shared__ float r1[256], r2[256];
  r1[tid] = v0 + v1; r2[tid] = v0 * v0 + v1 * v1;
  __syncthreads();
  for (int st = 128; st > 0; st >>= 1) {
    if (tid < st) { r1[tid] += r1[tid + st]; r2[tid] += r2[tid + st]; }
    __syncthreads();
  }
  float mean = r1[0] * (1.f / 512.f);
  float var  = r2[0] * (1.f / 512.f) - mean * mean;
  float rinv = rsqrtf(var + 1e-5f);
  int ro = permute ? ((r & ~1023) | ((r & 31) << 5) | ((r >> 5) & 31)) : r;
  size_t ob = (size_t)ro * 512;
  float y0 = (v0 - mean) * rinv * g[tid * 2]     + be[tid * 2];
  float y1 = (v1 - mean) * rinv * g[tid * 2 + 1] + be[tid * 2 + 1];
  outF[ob + tid * 2] = y0; outF[ob + tid * 2 + 1] = y1;
  if (outB) { outB[ob + tid * 2] = (bf16)y0; outB[ob + tid * 2 + 1] = (bf16)y1; }
}

// ---------------------------------------------------------------------------
// Tiled bf16 WMMA GEMM: C[M,N] = A[M,K](bf16) * Wt[N,K](bf16, pre-transposed)
// 256 threads = 8 waves; block tile 128x128, wave tile 64x32, BK=32.
// A/B tiles staged by the Tensor Data Mover, double-buffered in LDS so that
// TDM(k+1) overlaps WMMA(k). Wave 0 issues TDM and waits TENSORcnt; a
// workgroup barrier publishes the tile to all waves.
// FLAGS: 1=store f32, 2=store bf16, 4=GELU, 8=add residual f32.
// ---------------------------------------------------------------------------
template <int F>
__global__ __launch_bounds__(256) void gemm_bf16_k(
    const bf16* __restrict__ A, const bf16* __restrict__ Wt,
    const float* __restrict__ bias, const float* __restrict__ resid,
    float* __restrict__ outF, bf16* __restrict__ outB,
    int M, int N, int K) {
  __shared__ __align__(16) bf16 As[2][128 * 32];
  __shared__ __align__(16) bf16 Bs[2][128 * 32];
  const int bm = blockIdx.y * 128, bn = blockIdx.x * 128;
  const int tid = threadIdx.x;
  const int w = tid >> 5, l = tid & 31;
  const int wm = (w >> 2) * 64, wn = (w & 3) * 32;
  const int half = l >> 4, n16 = l & 15;
  const v8f vz = {0.f, 0.f, 0.f, 0.f, 0.f, 0.f, 0.f, 0.f};
  v8f acc[4][2];
#pragma unroll
  for (int mt = 0; mt < 4; ++mt)
#pragma unroll
    for (int nt = 0; nt < 2; ++nt) acc[mt][nt] = vz;

  const unsigned aLds[2] = {lds_lo32(&As[0][0]), lds_lo32(&As[1][0])};
  const unsigned bLds[2] = {lds_lo32(&Bs[0][0]), lds_lo32(&Bs[1][0])};
  const unsigned mRem = (unsigned)(M - bm);       // TDM zero-fills rows >= mRem
  const int kTiles = K >> 5;

  if (w == 0) {                                    // prologue: tile 0 in flight
    tdm_load_tile_2d(aLds[0], A + (size_t)bm * K, (unsigned)K, mRem,
                     32u, 128u, (unsigned long long)K);
    tdm_load_tile_2d(bLds[0], Wt + (size_t)bn * K, (unsigned)K, 128u,
                     32u, 128u, (unsigned long long)K);
  }

  for (int kt = 0; kt < kTiles; ++kt) {
    const int buf = kt & 1;
    if (w == 0) __builtin_amdgcn_s_wait_tensorcnt(0);
    __syncthreads();                               // publish tile kt
    if (w == 0 && kt + 1 < kTiles) {               // start tile kt+1 during compute
      const int k0n = (kt + 1) << 5;
      tdm_load_tile_2d(aLds[buf ^ 1], A + (size_t)bm * K + k0n,
                       (unsigned)(K - k0n), mRem, 32u, 128u, (unsigned long long)K);
      tdm_load_tile_2d(bLds[buf ^ 1], Wt + (size_t)bn * K + k0n,
                       (unsigned)(K - k0n), 128u, 32u, 128u, (unsigned long long)K);
    }
    const bf16* Ab = As[buf];
    const bf16* Bb = Bs[buf];
    AFrag a[4], b[2];
#pragma unroll
    for (int mt = 0; mt < 4; ++mt) {
      const bf16* p = Ab + (wm + mt * 16 + n16) * 32;
      a[mt].q[0] = *(const u32x4*)(p + 8 * half);        // K 8g..8g+7
      a[mt].q[1] = *(const u32x4*)(p + 16 + 8 * half);   // K 16+8g..16+8g+7
    }
#pragma unroll
    for (int nt = 0; nt < 2; ++nt) {
      const bf16* p = Bb + (wn + nt * 16 + n16) * 32;
      b[nt].q[0] = *(const u32x4*)(p + 8 * half);
      b[nt].q[1] = *(const u32x4*)(p + 16 + 8 * half);
    }
#pragma unroll
    for (int mt = 0; mt < 4; ++mt)
#pragma unroll
      for (int nt = 0; nt < 2; ++nt)
        acc[mt][nt] = __builtin_amdgcn_wmma_f32_16x16x32_bf16(
            false, a[mt].v, false, b[nt].v, (short)0, acc[mt][nt], false, false);
    __syncthreads();                               // done reading buf
  }

  // Epilogue (C layout: VGPR j -> M = j + 8*half, N = lane&15)
#pragma unroll
  for (int mt = 0; mt < 4; ++mt)
#pragma unroll
    for (int nt = 0; nt < 2; ++nt) {
      const int col = bn + wn + nt * 16 + n16;
      const float bv = bias[col];
#pragma unroll
      for (int j = 0; j < 8; ++j) {
        const int row = bm + wm + mt * 16 + j + 8 * half;
        if (row < M) {
          float v = acc[mt][nt][j] + bv;
          if (F & 8) v += resid[(size_t)row * N + col];
          if (F & 4) v = gelu_exact(v);
          if (F & 1) outF[(size_t)row * N + col] = v;
          if (F & 2) outB[(size_t)row * N + col] = (bf16)v;
        }
      }
    }
}

// ---------------------------------------------------------------------------
// Generic small attention core: one wave per (group, head).
// Lq,Lk in {16,32}; DH=64; S = QK^T/8 -> masked softmax -> O = P V (bf16 out).
// qValid: output rows >= qValid are written as zero (router FACTOR=10 pad).
// kValid: softmax only over keys < kValid (buf FACTOR=10 pad).
// ---------------------------------------------------------------------------
__global__ __launch_bounds__(32) void attn_k(
    const bf16* __restrict__ Q, const bf16* __restrict__ K,
    const bf16* __restrict__ V, bf16* __restrict__ O,
    int qRowsPerG, int kRowsPerG, int oRowsPerG,
    int Lq, int Lk, int qValid, int kValid, int qUseSeg) {
  __shared__ __align__(16) bf16  Qs[32 * 64];
  __shared__ __align__(16) bf16  Ks[32 * 64];
  __shared__ __align__(16) bf16  Vt[64 * 32];   // transposed: [dh][seq]
  __shared__ __align__(16) float Ss[32 * 32];
  __shared__ __align__(16) bf16  Pbf[32 * 32];
  const int g = blockIdx.x >> 3, h = blockIdx.x & 7;
  const int qg = qUseSeg ? (g & 31) : g;        // router Q is shared across batch
  const size_t qRow0 = (size_t)qg * qRowsPerG;
  const size_t kRow0 = (size_t)g * kRowsPerG;
  const size_t oRow0 = (size_t)g * oRowsPerG;
  const int l = threadIdx.x;
  const int half = l >> 4, n16 = l & 15;
  const u32x4 z4 = {0u, 0u, 0u, 0u};
  const v8f vz = {0.f, 0.f, 0.f, 0.f, 0.f, 0.f, 0.f, 0.f};

  // Stage Q, K rows and V transposed (zero-pad beyond Lq/Lk)
  {
    u32x4* dq = (u32x4*)(Qs + l * 64);
    if (l < Lq) {
      const u32x4* sq = (const u32x4*)(Q + (qRow0 + l) * 512 + h * 64);
#pragma unroll
      for (int i = 0; i < 8; ++i) dq[i] = sq[i];
    } else {
#pragma unroll
      for (int i = 0; i < 8; ++i) dq[i] = z4;
    }
    u32x4* dk = (u32x4*)(Ks + l * 64);
    if (l < Lk) {
      const u32x4* sk = (const u32x4*)(K + (kRow0 + l) * 512 + h * 64);
#pragma unroll
      for (int i = 0; i < 8; ++i) dk[i] = sk[i];
      const bf16* vs = V + (kRow0 + l) * 512 + h * 64;
      for (int d = 0; d < 64; ++d) Vt[d * 32 + l] = vs[d];
    } else {
#pragma unroll
      for (int i = 0; i < 8; ++i) dk[i] = z4;
      for (int d = 0; d < 64; ++d) Vt[d * 32 + l] = (bf16)0.0f;
    }
  }
  __syncthreads();

  const int nMt = Lq >> 4, nNt = Lk >> 4;
  // S = Q K^T * (1/sqrt(64))
  for (int mt = 0; mt < nMt; ++mt)
    for (int nt = 0; nt < nNt; ++nt) {
      v8f acc = vz;
#pragma unroll
      for (int kc = 0; kc < 2; ++kc) {
        AFrag a, b;
        const bf16* pa = Qs + (mt * 16 + n16) * 64 + kc * 32;
        a.q[0] = *(const u32x4*)(pa + 8 * half);
        a.q[1] = *(const u32x4*)(pa + 16 + 8 * half);
        const bf16* pb = Ks + (nt * 16 + n16) * 64 + kc * 32;
        b.q[0] = *(const u32x4*)(pb + 8 * half);
        b.q[1] = *(const u32x4*)(pb + 16 + 8 * half);
        acc = __builtin_amdgcn_wmma_f32_16x16x32_bf16(
            false, a.v, false, b.v, (short)0, acc, false, false);
      }
#pragma unroll
      for (int j = 0; j < 8; ++j)
        Ss[(mt * 16 + j + 8 * half) * 32 + nt * 16 + n16] = acc[j] * 0.125f;
    }
  __syncthreads();

  // Row softmax over valid keys; write bf16 probs padded with zeros to K=32
  if (l < Lq) {
    float mx = -1e30f;
    for (int c = 0; c < kValid; ++c) mx = fmaxf(mx, Ss[l * 32 + c]);
    float sum = 0.f;
    for (int c = 0; c < kValid; ++c) {
      float e = __expf(Ss[l * 32 + c] - mx);
      sum += e; Ss[l * 32 + c] = e;
    }
    float inv = 1.f / sum;
    for (int c = 0; c < 32; ++c)
      Pbf[l * 32 + c] = (c < kValid) ? (bf16)(Ss[l * 32 + c] * inv) : (bf16)0.0f;
  }
  __syncthreads();

  // O = P V  (K padded to 32 with zeros)
  for (int mt = 0; mt < nMt; ++mt)
#pragma unroll
    for (int nt = 0; nt < 4; ++nt) {
      v8f acc = vz;
      AFrag a, b;
      const bf16* pa = Pbf + (mt * 16 + n16) * 32;
      a.q[0] = *(const u32x4*)(pa + 8 * half);
      a.q[1] = *(const u32x4*)(pa + 16 + 8 * half);
      const bf16* pb = Vt + (nt * 16 + n16) * 32;
      b.q[0] = *(const u32x4*)(pb + 8 * half);
      b.q[1] = *(const u32x4*)(pb + 16 + 8 * half);
      acc = __builtin_amdgcn_wmma_f32_16x16x32_bf16(
          false, a.v, false, b.v, (short)0, acc, false, false);
#pragma unroll
      for (int j = 0; j < 8; ++j) {
        int mrow = mt * 16 + j + 8 * half;
        float v = (mrow < qValid) ? acc[j] : 0.f;
        O[(oRow0 + mrow) * 512 + h * 64 + nt * 16 + n16] = (bf16)v;
      }
    }
}

// ---------------------------------------------------------------------------
// Host orchestration
// ---------------------------------------------------------------------------
extern "C" void kernel_launch(void* const* d_in, const int* in_sizes, int n_in,
                              void* d_out, int out_size, void* d_ws, size_t ws_size,
                              hipStream_t stream) {
  (void)in_sizes; (void)n_in; (void)out_size; (void)ws_size;
  const float* x        = (const float*)d_in[0];
  const float* merge_g  = (const float*)d_in[1];
  const float* merge_be = (const float*)d_in[2];
  const float* merge_W  = (const float*)d_in[3];
  const float* merge_b  = (const float*)d_in[4];
  const float* t_qW = (const float*)d_in[5];  const float* t_qb = (const float*)d_in[6];
  const float* t_kW = (const float*)d_in[7];  const float* t_kb = (const float*)d_in[8];
  const float* t_vW = (const float*)d_in[9];  const float* t_vb = (const float*)d_in[10];
  const float* t_oW = (const float*)d_in[11]; const float* t_ob = (const float*)d_in[12];
  const float* s_qW = (const float*)d_in[13]; const float* s_qb = (const float*)d_in[14];
  const float* s_kW = (const float*)d_in[15]; const float* s_kb = (const float*)d_in[16];
  const float* s_vW = (const float*)d_in[17]; const float* s_vb = (const float*)d_in[18];
  const float* s_oW = (const float*)d_in[19]; const float* s_ob = (const float*)d_in[20];
  const float* r_qW = (const float*)d_in[21]; const float* r_qb = (const float*)d_in[22];
  const float* r_kW = (const float*)d_in[23]; const float* r_kb = (const float*)d_in[24];
  const float* r_vW = (const float*)d_in[25]; const float* r_vb = (const float*)d_in[26];
  const float* r_oW = (const float*)d_in[27]; const float* r_ob = (const float*)d_in[28];
  const float* router = (const float*)d_in[29];
  const float* n1g = (const float*)d_in[30]; const float* n1b = (const float*)d_in[31];
  const float* n2g = (const float*)d_in[32]; const float* n2b = (const float*)d_in[33];
  const float* n3g = (const float*)d_in[34]; const float* n3b = (const float*)d_in[35];
  const float* n4g = (const float*)d_in[36]; const float* n4b = (const float*)d_in[37];
  const float* m1W1 = (const float*)d_in[38]; const float* m1b1 = (const float*)d_in[39];
  const float* m1W2 = (const float*)d_in[40]; const float* m1b2 = (const float*)d_in[41];
  const float* m2W1 = (const float*)d_in[42]; const float* m2b1 = (const float*)d_in[43];
  const float* m2W2 = (const float*)d_in[44]; const float* m2b2 = (const float*)d_in[45];
  float* out = (float*)d_out;

  char* ws = (char*)d_ws;
  size_t off = 0;
  auto alloc = [&](size_t bytes) -> char* {
    char* p = ws + off; off = (off + bytes + 255) & ~(size_t)255; return p;
  };

  // bf16 transposed weights
  bf16* wT_merge = (bf16*)alloc((size_t)1024 * 512 * 2);
  bf16* wT_tq = (bf16*)alloc((size_t)512 * 512 * 2);
  bf16* wT_tk = (bf16*)alloc((size_t)512 * 512 * 2);
  bf16* wT_tv = (bf16*)alloc((size_t)512 * 512 * 2);
  bf16* wT_to = (bf16*)alloc((size_t)512 * 512 * 2);
  bf16* wT_sq = (bf16*)alloc((size_t)512 * 512 * 2);
  bf16* wT_sk = (bf16*)alloc((size_t)512 * 512 * 2);
  bf16* wT_sv = (bf16*)alloc((size_t)512 * 512 * 2);
  bf16* wT_so = (bf16*)alloc((size_t)512 * 512 * 2);
  bf16* wT_rq = (bf16*)alloc((size_t)512 * 512 * 2);
  bf16* wT_rk = (bf16*)alloc((size_t)512 * 512 * 2);
  bf16* wT_rv = (bf16*)alloc((size_t)512 * 512 * 2);
  bf16* wT_ro = (bf16*)alloc((size_t)512 * 512 * 2);
  bf16* wT_m1a = (bf16*)alloc((size_t)512 * 1024 * 2);
  bf16* wT_m1b = (bf16*)alloc((size_t)1024 * 512 * 2);
  bf16* wT_m2a = (bf16*)alloc((size_t)512 * 1024 * 2);
  bf16* wT_m2b = (bf16*)alloc((size_t)1024 * 512 * 2);
  bf16* routerPad = (bf16*)alloc((size_t)512 * 512 * 2);
  bf16* Qr  = (bf16*)alloc((size_t)512 * 512 * 2);
  // activation slots (liveness-based reuse)
  bf16* sA   = (bf16*)alloc((size_t)32768 * 1024 * 2); // xmln / mlp hidden
  bf16* sB   = (bf16*)alloc((size_t)32768 * 512 * 2);  // xm_bf / t1_bf / ds_bf / e1_bf
  bf16* sQ   = (bf16*)alloc((size_t)32768 * 512 * 2);  // Q / Qd
  bf16* sK   = (bf16*)alloc((size_t)32768 * 512 * 2);  // K / Ks / Kr
  bf16* sV   = (bf16*)alloc((size_t)32768 * 512 * 2);  // V / Vs / Vr
  bf16* sATT = (bf16*)alloc((size_t)32768 * 512 * 2);  // attn out / sbuf / att2
  bf16* sBUF = (bf16*)alloc((size_t)16384 * 512 * 2);  // router buffer (padded 16)
  float* R1  = (float*)alloc((size_t)32768 * 512 * 4); // xm / t1 / ds / e1 (f32)
  float* R2  = (float*)alloc((size_t)32768 * 512 * 4); // t0 / t2 / e0 / e2 (f32)

  auto cvt = [&](const float* W, bf16* Wt, int K, int N) {
    int n = K * N;
    wcvt_k<<<dim3((n + 255) / 256), dim3(256), 0, stream>>>(W, Wt, K, N);
  };
  cvt(merge_W, wT_merge, 1024, 512);
  cvt(t_qW, wT_tq, 512, 512); cvt(t_kW, wT_tk, 512, 512);
  cvt(t_vW, wT_tv, 512, 512); cvt(t_oW, wT_to, 512, 512);
  cvt(s_qW, wT_sq, 512, 512); cvt(s_kW, wT_sk, 512, 512);
  cvt(s_vW, wT_sv, 512, 512); cvt(s_oW, wT_so, 512, 512);
  cvt(r_qW, wT_rq, 512, 512); cvt(r_kW, wT_rk, 512, 512);
  cvt(r_vW, wT_rv, 512, 512); cvt(r_oW, wT_ro, 512, 512);
  cvt(m1W1, wT_m1a, 512, 1024); cvt(m1W2, wT_m1b, 1024, 512);
  cvt(m2W1, wT_m2a, 512, 1024); cvt(m2W2, wT_m2b, 1024, 512);
  router_pad_k<<<dim3(262144 / 256), dim3(256), 0, stream>>>(router, routerPad);

  const dim3 blk256(256), blk32(32);
  const dim3 gM512(4, 256);    // N=512 x M=32768
  const dim3 gDFF(8, 256);     // N=1024 x M=32768
  const dim3 gBuf(4, 128);     // N=512 x M=16384

  // --- SegMerging ---
  merge_ln_k<<<dim3(32768), blk256, 0, stream>>>(x, merge_g, merge_be, sA);
  gemm_bf16_k<3><<<gM512, blk256, 0, stream>>>(sA, wT_merge, merge_b, nullptr, R1, sB, 32768, 512, 1024);

  // --- Stage 1: time attention ---
  gemm_bf16_k<2><<<gM512, blk256, 0, stream>>>(sB, wT_tq, t_qb, nullptr, nullptr, sQ, 32768, 512, 512);
  gemm_bf16_k<2><<<gM512, blk256, 0, stream>>>(sB, wT_tk, t_kb, nullptr, nullptr, sK, 32768, 512, 512);
  gemm_bf16_k<2><<<gM512, blk256, 0, stream>>>(sB, wT_tv, t_vb, nullptr, nullptr, sV, 32768, 512, 512);
  attn_k<<<dim3(8192), blk32, 0, stream>>>(sQ, sK, sV, sATT, 32, 32, 32, 32, 32, 32, 32, 0);
  gemm_bf16_k<9><<<gM512, blk256, 0, stream>>>(sATT, wT_to, t_ob, R1, R2, nullptr, 32768, 512, 512);
  ln512_k<<<dim3(32768), blk256, 0, stream>>>(R2, n1g, n1b, R1, sB, 0);
  gemm_bf16_k<6><<<gDFF, blk256, 0, stream>>>(sB, wT_m1a, m1b1, nullptr, nullptr, sA, 32768, 1024, 512);
  gemm_bf16_k<9><<<gM512, blk256, 0, stream>>>(sA, wT_m1b, m1b2, R1, R2, nullptr, 32768, 512, 1024);
  // LN n2 with t -> ds permute: R1 = ds_f32, sB = ds_bf16
  ln512_k<<<dim3(32768), blk256, 0, stream>>>(R2, n2g, n2b, R1, sB, 1);

  // --- Stage 2: router attention ---
  gemm_bf16_k<2><<<dim3(4, 4), blk256, 0, stream>>>(routerPad, wT_sq, s_qb, nullptr, nullptr, Qr, 512, 512, 512);
  gemm_bf16_k<2><<<gM512, blk256, 0, stream>>>(sB, wT_sk, s_kb, nullptr, nullptr, sK, 32768, 512, 512);
  gemm_bf16_k<2><<<gM512, blk256, 0, stream>>>(sB, wT_sv, s_vb, nullptr, nullptr, sV, 32768, 512, 512);
  // send: router(10 padded to 16) queries over 32 channels
  attn_k<<<dim3(8192), blk32, 0, stream>>>(Qr, sK, sV, sATT, 16, 32, 16, 16, 32, 10, 32, 1);
  gemm_bf16_k<2><<<gBuf, blk256, 0, stream>>>(sATT, wT_so, s_ob, nullptr, nullptr, sBUF, 16384, 512, 512);
  gemm_bf16_k<2><<<gBuf, blk256, 0, stream>>>(sBUF, wT_rk, r_kb, nullptr, nullptr, sK, 16384, 512, 512);
  gemm_bf16_k<2><<<gBuf, blk256, 0, stream>>>(sBUF, wT_rv, r_vb, nullptr, nullptr, sV, 16384, 512, 512);
  gemm_bf16_k<2><<<gM512, blk256, 0, stream>>>(sB, wT_rq, r_qb, nullptr, nullptr, sQ, 32768, 512, 512);
  // recv: 32 channel queries over 10 valid buffer keys (padded to 16)
  attn_k<<<dim3(8192), blk32, 0, stream>>>(sQ, sK, sV, sATT, 32, 16, 32, 32, 16, 32, 10, 0);
  gemm_bf16_k<9><<<gM512, blk256, 0, stream>>>(sATT, wT_ro, r_ob, R1, R2, nullptr, 32768, 512, 512);
  ln512_k<<<dim3(32768), blk256, 0, stream>>>(R2, n3g, n3b, R1, sB, 0);
  gemm_bf16_k<6><<<gDFF, blk256, 0, stream>>>(sB, wT_m2a, m2b1, nullptr, nullptr, sA, 32768, 1024, 512);
  gemm_bf16_k<9><<<gM512, blk256, 0, stream>>>(sA, wT_m2b, m2b2, R1, R2, nullptr, 32768, 512, 1024);
  // LN n4 with ds -> output permute, straight into d_out (f32)
  ln512_k<<<dim3(32768), blk256, 0, stream>>>(R2, n4g, n4b, out, nullptr, 1);
}
